// BahdanauAttnDecoderRNN_53506702574088
// MI455X (gfx1250) — matrix operand order repaired and checked by reading, not compiled, not run on hardware
//
#include <hip/hip_runtime.h>
#include <hip/hip_bf16.h>
#include <math.h>

// Problem dims
#define S_LEN  128
#define BATCH  64
#define HID    512
#define VOCAB  100
#define NSTEPS 31   // T-1

typedef __attribute__((ext_vector_type(16))) _Float16 v16h;
typedef __attribute__((ext_vector_type(8)))  _Float16 v8h;
typedef __attribute__((ext_vector_type(8)))  float    v8f;

// ---------------------------------------------------------------------------
// WMMA helpers (documented gfx1250 16-bit operand layouts, wave32)
// ---------------------------------------------------------------------------
__device__ __forceinline__ v8f wmma16(v16h a, v16h b, v8f c) {
  return __builtin_amdgcn_wmma_f32_16x16x32_f16(
      /*neg_a=*/false, a, /*neg_b=*/false, b,
      /*c_mod=*/(short)0, c, /*reuse_a=*/false, /*reuse_b=*/false);
}

// A: row-major [M][K] f16. Lane l<16 holds M=l, K = k0+{0..7, 16..23};
// lane l>=16 holds M=l-16, K = k0+{8..15, 24..31}.
__device__ __forceinline__ v16h load_a_tile(const _Float16* A, int lda,
                                            int m0, int k0, int lane) {
  int l  = lane & 15;
  int hi = (lane >> 4) & 1;
  const _Float16* row = A + (size_t)(m0 + l) * lda + k0 + hi * 8;
  v8h c0 = *(const v8h*)(row);
  v8h c1 = *(const v8h*)(row + 16);
  v16h a;
#pragma unroll
  for (int i = 0; i < 8; ++i) { a[i] = c0[i]; a[i + 8] = c1[i]; }
  return a;
}

// B: pre-transposed weight [K][N] f16. Lane l holds column N=n0+(l&15),
// K = k0+8*(l>>4)+{0..7, 16..23} (same k order as A).
__device__ __forceinline__ v16h load_b_tile(const _Float16* Bt, int ldb,
                                            int k0, int n0, int lane) {
  int n  = n0 + (lane & 15);
  int hi = (lane >> 4) & 1;
  const _Float16* col = Bt + (size_t)(k0 + hi * 8) * ldb + n;
  v16h b;
#pragma unroll
  for (int i = 0; i < 8; ++i) {
    b[i]     = col[(size_t)i * ldb];
    b[i + 8] = col[(size_t)(i + 16) * ldb];
  }
  return b;
}

// ---------------------------------------------------------------------------
// One-time per launch: transpose f32 weight [N][K] -> f16 [K][N]
// ---------------------------------------------------------------------------
__global__ void transpose_to_f16(const float* __restrict__ src,
                                 _Float16* __restrict__ dst, int N, int K) {
  int idx = blockIdx.x * blockDim.x + threadIdx.x;
  if (idx >= N * K) return;
  int n = idx / K;
  int k = idx - n * K;
  dst[(size_t)k * N + n] = (_Float16)src[idx];
}

__global__ void init_h(float* __restrict__ h0, _Float16* __restrict__ h0f) {
  int idx = blockIdx.x * blockDim.x + threadIdx.x;
  if (idx < BATCH * HID) { h0[idx] = 0.0f; h0f[idx] = (_Float16)0.0f; }
}

// ---------------------------------------------------------------------------
// K1: scores[s,b] = tanh(enc[s,b,:] + h[b,:]) . w_vat + b_vat    (one wave per pair)
// ---------------------------------------------------------------------------
__global__ void attn_scores(const float* __restrict__ enc,
                            const float* __restrict__ h,
                            const float* __restrict__ w_vat,
                            const float* __restrict__ b_vat,
                            float* __restrict__ score) {
  int pair = blockIdx.x * 8 + (threadIdx.x >> 5);
  int lane = threadIdx.x & 31;
  int b = pair & (BATCH - 1);
  int s = pair >> 6;
  const float* e  = enc + ((size_t)s * BATCH + b) * HID;
  const float* hb = h + (size_t)b * HID;
  float acc = 0.0f;
  for (int k = lane; k < HID; k += 32)
    acc += tanhf(e[k] + hb[k]) * w_vat[k];
#pragma unroll
  for (int o = 16; o > 0; o >>= 1) acc += __shfl_xor(acc, o, 32);
  if (lane == 0) score[s * BATCH + b] = acc + b_vat[0];
}

// ---------------------------------------------------------------------------
// K2: per-batch softmax over S, ctx = sum_s p[s]*enc[s,b,:], and build
//     cat_f16[b] = [ W_emb[sym] | ctx ]  (f16 for the WMMA combine GEMM)
// ---------------------------------------------------------------------------
__global__ void softmax_ctx_cat(const float* __restrict__ enc,
                                const float* __restrict__ score,
                                const float* __restrict__ W_emb,
                                const int* __restrict__ tgt, int t,
                                _Float16* __restrict__ cat_f16) {
  __shared__ float p[S_LEN];
  __shared__ float red[S_LEN];
  int b = blockIdx.x;
  int tid = threadIdx.x;

  if (tid < S_LEN) { float v = score[tid * BATCH + b]; p[tid] = v; red[tid] = v; }
  __syncthreads();
  for (int st = S_LEN / 2; st > 0; st >>= 1) {
    if (tid < st) red[tid] = fmaxf(red[tid], red[tid + st]);
    __syncthreads();
  }
  float m = red[0];
  __syncthreads();
  float e = 0.0f;
  if (tid < S_LEN) { e = expf(p[tid] - m); red[tid] = e; }
  __syncthreads();
  for (int st = S_LEN / 2; st > 0; st >>= 1) {
    if (tid < st) red[tid] += red[tid + st];
    __syncthreads();
  }
  float inv = 1.0f / red[0];
  __syncthreads();
  if (tid < S_LEN) p[tid] = e * inv;
  __syncthreads();

  // ctx into cat[:,512:1024]
  for (int hc = tid; hc < HID; hc += blockDim.x) {
    float acc = 0.0f;
#pragma unroll 4
    for (int s = 0; s < S_LEN; ++s)
      acc += p[s] * enc[((size_t)s * BATCH + b) * HID + hc];
    cat_f16[(size_t)b * (2 * HID) + HID + hc] = (_Float16)acc;
  }
  // embedding gather into cat[:,0:512]
  int sym = tgt[t * BATCH + b];
  for (int hc = tid; hc < HID; hc += blockDim.x)
    cat_f16[(size_t)b * (2 * HID) + hc] = (_Float16)W_emb[(size_t)sym * HID + hc];
}

// ---------------------------------------------------------------------------
// K3a: x = relu(cat @ W_comb^T + b_comb)   M=64 N=512 K=1024, WMMA f16->f32
// ---------------------------------------------------------------------------
__global__ void combine_gemm(const _Float16* __restrict__ cat_f16,
                             const _Float16* __restrict__ Wct,   // [1024][512]
                             const float* __restrict__ b_comb,
                             _Float16* __restrict__ x_f16) {
  int tile = blockIdx.x * 4 + (threadIdx.x >> 5);   // 128 tiles (4x32)
  int lane = threadIdx.x & 31;
  int mt = tile >> 5, nt = tile & 31;
  int m0 = mt * 16, n0 = nt * 16;
  v8f c = {};
  for (int kb = 0; kb < 2 * HID; kb += 32) {
    v16h a = load_a_tile(cat_f16, 2 * HID, m0, kb, lane);
    v16h b = load_b_tile(Wct, HID, kb, n0, lane);
    c = wmma16(a, b, c);
  }
  int n  = n0 + (lane & 15);
  int mb = m0 + ((lane >> 4) << 3);
  float bias = b_comb[n];
#pragma unroll
  for (int r = 0; r < 8; ++r) {
    float v = c[r] + bias;
    v = v > 0.0f ? v : 0.0f;
    x_f16[(size_t)(mb + r) * HID + n] = (_Float16)v;
  }
}

// ---------------------------------------------------------------------------
// K3b: fused GRU cell: gi = x@W_ih^T + b_ih, gh = h@W_hh^T + b_hh, gates,
//      h_new.  6 WMMA tile-chains per wave (r/z/n x input/hidden), K=512.
// ---------------------------------------------------------------------------
__global__ void gru_cell(const _Float16* __restrict__ x_f16,
                         const _Float16* __restrict__ h_f16,
                         const _Float16* __restrict__ Wiht,  // [512][1536]
                         const _Float16* __restrict__ Whht,  // [512][1536]
                         const float* __restrict__ b_ih,
                         const float* __restrict__ b_hh,
                         const float* __restrict__ h_cur,
                         float* __restrict__ h_next,
                         _Float16* __restrict__ h_next_f16) {
  int tile = blockIdx.x * 4 + (threadIdx.x >> 5);
  int lane = threadIdx.x & 31;
  int mt = tile >> 5, nt = tile & 31;
  int m0 = mt * 16, n0 = nt * 16;
  v8f ci0 = {}, ci1 = {}, ci2 = {}, ch0 = {}, ch1 = {}, ch2 = {};
  for (int kb = 0; kb < HID; kb += 32) {
    v16h ax = load_a_tile(x_f16, HID, m0, kb, lane);
    v16h ah = load_a_tile(h_f16, HID, m0, kb, lane);
    v16h bi0 = load_b_tile(Wiht, 3 * HID, kb, n0,            lane);
    v16h bi1 = load_b_tile(Wiht, 3 * HID, kb, n0 + HID,      lane);
    v16h bi2 = load_b_tile(Wiht, 3 * HID, kb, n0 + 2 * HID,  lane);
    ci0 = wmma16(ax, bi0, ci0);
    ci1 = wmma16(ax, bi1, ci1);
    ci2 = wmma16(ax, bi2, ci2);
    v16h bh0 = load_b_tile(Whht, 3 * HID, kb, n0,            lane);
    v16h bh1 = load_b_tile(Whht, 3 * HID, kb, n0 + HID,      lane);
    v16h bh2 = load_b_tile(Whht, 3 * HID, kb, n0 + 2 * HID,  lane);
    ch0 = wmma16(ah, bh0, ch0);
    ch1 = wmma16(ah, bh1, ch1);
    ch2 = wmma16(ah, bh2, ch2);
  }
  int n  = n0 + (lane & 15);
  int mb = m0 + ((lane >> 4) << 3);
  float bir = b_ih[n], biz = b_ih[n + HID], bin = b_ih[n + 2 * HID];
  float bhr = b_hh[n], bhz = b_hh[n + HID], bhn = b_hh[n + 2 * HID];
#pragma unroll
  for (int r = 0; r < 8; ++r) {
    int m = mb + r;
    float ir = ci0[r] + bir, iz = ci1[r] + biz, in = ci2[r] + bin;
    float hr = ch0[r] + bhr, hz = ch1[r] + bhz, hn = ch2[r] + bhn;
    float rg = 1.0f / (1.0f + expf(-(ir + hr)));
    float zg = 1.0f / (1.0f + expf(-(iz + hz)));
    float ng = tanhf(in + rg * hn);
    float ho = h_cur[(size_t)m * HID + n];
    float hv = (1.0f - zg) * ng + zg * ho;
    h_next[(size_t)m * HID + n] = hv;
    h_next_f16[(size_t)m * HID + n] = (_Float16)hv;
  }
}

// ---------------------------------------------------------------------------
// K4: logits = log_softmax(h_new @ W_out^T + b_out)  (V=100, VALU + LDS)
//     out layout: [B, T-1, V]
// ---------------------------------------------------------------------------
__global__ void logits_lsm(const float* __restrict__ h_new,
                           const float* __restrict__ W_out,
                           const float* __restrict__ b_out,
                           float* __restrict__ out, int t) {
  __shared__ float lg[128];
  __shared__ float red[128];
  int b = blockIdx.x, tid = threadIdx.x;
  float acc = -INFINITY;
  if (tid < VOCAB) {
    const float* hb = h_new + (size_t)b * HID;
    const float* w  = W_out + (size_t)tid * HID;
    float s = 0.0f;
#pragma unroll 4
    for (int k = 0; k < HID; ++k) s += hb[k] * w[k];
    acc = s + b_out[tid];
  }
  lg[tid] = acc; red[tid] = acc;
  __syncthreads();
  for (int st = 64; st > 0; st >>= 1) {
    if (tid < st) red[tid] = fmaxf(red[tid], red[tid + st]);
    __syncthreads();
  }
  float m = red[0];
  __syncthreads();
  float e = (tid < VOCAB) ? expf(lg[tid] - m) : 0.0f;
  red[tid] = e;
  __syncthreads();
  for (int st = 64; st > 0; st >>= 1) {
    if (tid < st) red[tid] += red[tid + st];
    __syncthreads();
  }
  float lse = logf(red[0]);
  if (tid < VOCAB)
    out[((size_t)b * NSTEPS + t) * VOCAB + tid] = lg[tid] - m - lse;
}

// ---------------------------------------------------------------------------
// Host-side launch
// ---------------------------------------------------------------------------
static inline char* ws_alloc(char*& cur, size_t bytes) {
  char* p = cur;
  cur += (bytes + 255) & ~(size_t)255;
  return p;
}

extern "C" void kernel_launch(void* const* d_in, const int* in_sizes, int n_in,
                              void* d_out, int out_size, void* d_ws, size_t ws_size,
                              hipStream_t stream) {
  (void)in_sizes; (void)n_in; (void)out_size; (void)ws_size;
  const float* enc    = (const float*)d_in[0];
  const float* W_emb  = (const float*)d_in[1];
  const float* W_comb = (const float*)d_in[2];
  const float* b_comb = (const float*)d_in[3];
  const float* W_ih   = (const float*)d_in[4];
  const float* W_hh   = (const float*)d_in[5];
  const float* b_ih   = (const float*)d_in[6];
  const float* b_hh   = (const float*)d_in[7];
  const float* W_out  = (const float*)d_in[8];
  const float* b_out  = (const float*)d_in[9];
  const float* w_vat  = (const float*)d_in[10];
  const float* b_vat  = (const float*)d_in[11];
  const int*   tgt    = (const int*)d_in[12];
  float* out = (float*)d_out;

  char* cur = (char*)d_ws;
  float*    hbuf0 = (float*)   ws_alloc(cur, BATCH * HID * sizeof(float));
  float*    hbuf1 = (float*)   ws_alloc(cur, BATCH * HID * sizeof(float));
  _Float16* hf0   = (_Float16*)ws_alloc(cur, BATCH * HID * sizeof(_Float16));
  _Float16* hf1   = (_Float16*)ws_alloc(cur, BATCH * HID * sizeof(_Float16));
  float*    score = (float*)   ws_alloc(cur, S_LEN * BATCH * sizeof(float));
  _Float16* catf  = (_Float16*)ws_alloc(cur, BATCH * 2 * HID * sizeof(_Float16));
  _Float16* xf    = (_Float16*)ws_alloc(cur, BATCH * HID * sizeof(_Float16));
  _Float16* Wct   = (_Float16*)ws_alloc(cur, (size_t)2 * HID * HID * sizeof(_Float16));
  _Float16* Wiht  = (_Float16*)ws_alloc(cur, (size_t)HID * 3 * HID * sizeof(_Float16));
  _Float16* Whht  = (_Float16*)ws_alloc(cur, (size_t)HID * 3 * HID * sizeof(_Float16));
  float* hb[2]    = { hbuf0, hbuf1 };
  _Float16* hf[2] = { hf0, hf1 };

  // One-time per launch: weight transpose+convert, h0 = 0
  transpose_to_f16<<<(HID * 2 * HID + 255) / 256, 256, 0, stream>>>(W_comb, Wct, HID, 2 * HID);
  transpose_to_f16<<<(3 * HID * HID + 255) / 256, 256, 0, stream>>>(W_ih, Wiht, 3 * HID, HID);
  transpose_to_f16<<<(3 * HID * HID + 255) / 256, 256, 0, stream>>>(W_hh, Whht, 3 * HID, HID);
  init_h<<<(BATCH * HID + 255) / 256, 256, 0, stream>>>(hb[0], hf[0]);

  for (int t = 0; t < NSTEPS; ++t) {
    int c = t & 1, n = c ^ 1;
    attn_scores<<<S_LEN * BATCH / 8, 256, 0, stream>>>(enc, hb[c], w_vat, b_vat, score);
    softmax_ctx_cat<<<BATCH, 256, 0, stream>>>(enc, score, W_emb, tgt, t, catf);
    combine_gemm<<<32, 128, 0, stream>>>(catf, Wct, b_comb, xf);
    gru_cell<<<32, 128, 0, stream>>>(xf, hf[c], Wiht, Whht, b_ih, b_hh,
                                     hb[c], hb[n], hf[n]);
    logits_lsm<<<BATCH, 128, 0, stream>>>(hb[n], W_out, b_out, out, t);
  }
}
